// DyConv_74904229642549
// MI455X (gfx1250) — matrix operand order repaired and checked by reading, not compiled, not run on hardware
//
#include <hip/hip_runtime.h>
#include <hip/hip_bf16.h>
#include <math.h>

typedef __attribute__((ext_vector_type(16))) __bf16        v16bf;
typedef __attribute__((ext_vector_type(8)))  float         v8f;
typedef __attribute__((ext_vector_type(4)))  unsigned int  v4u;

union FragU { v4u u[2]; v16bf v; };

__device__ __forceinline__ unsigned short f2bf(float f) {
  unsigned u = __float_as_uint(f);
  unsigned r = u + 0x7FFFu + ((u >> 16) & 1u);   // round-to-nearest-even
  return (unsigned short)(r >> 16);
}

#define B_      16
#define C1_     64
#define C2_     64
#define E_      4
#define R_      4
#define H_      160
#define W_      160
#define KK_     576     // 9 * 64, k = (kh*3+kw)*64 + c1
#define TILE_H  4
#define TILE_W  32
#define HALO_H  6
#define HALO_W  34

// ---------------- Stage 1: global average pool per (b, c1) ----------------
__global__ __launch_bounds__(256) void pool_kernel(const float* __restrict__ x,
                                                   float* __restrict__ pooled) {
  const int idx = blockIdx.x;                       // b*C1 + c
  const float* p = x + (size_t)idx * (H_ * W_);
  float s = 0.f;
  for (int i = threadIdx.x; i < H_ * W_; i += 256) s += p[i];
  for (int o = 16; o > 0; o >>= 1) s += __shfl_down(s, o, 32);
  __shared__ float red[8];
  const int lane = threadIdx.x & 31, wave = threadIdx.x >> 5;
  if (lane == 0) red[wave] = s;
  __syncthreads();
  if (threadIdx.x == 0) {
    float t = 0.f;
    #pragma unroll
    for (int i = 0; i < 8; ++i) t += red[i];
    pooled[idx] = t * (1.0f / (H_ * W_));
  }
}

// --------- Stage 2: attention MLP + softmax, folded BN coefficients --------
__global__ void attn_bn_kernel(const float* __restrict__ pooled,   // [B,C1]
                               const float* __restrict__ w1,       // [R,C1]
                               const float* __restrict__ w2,       // [E,R]
                               const float* __restrict__ gamma,
                               const float* __restrict__ beta,
                               const float* __restrict__ mean,
                               const float* __restrict__ var,
                               float* __restrict__ attn,           // [B,E]
                               float* __restrict__ bnscale,        // [C2]
                               float* __restrict__ bnbias) {       // [C2]
  const int tid = threadIdx.x;
  if (tid < B_) {
    float h[R_];
    #pragma unroll
    for (int r = 0; r < R_; ++r) {
      float s = 0.f;
      for (int c = 0; c < C1_; ++c) s += pooled[tid * C1_ + c] * w1[r * C1_ + c];
      h[r] = fmaxf(s, 0.f);
    }
    float lg[E_]; float m = -1e30f;
    #pragma unroll
    for (int e = 0; e < E_; ++e) {
      float s = 0.f;
      #pragma unroll
      for (int r = 0; r < R_; ++r) s += h[r] * w2[e * R_ + r];
      lg[e] = s; m = fmaxf(m, s);
    }
    float den = 0.f;
    #pragma unroll
    for (int e = 0; e < E_; ++e) { lg[e] = __expf(lg[e] - m); den += lg[e]; }
    const float inv = 1.f / den;
    #pragma unroll
    for (int e = 0; e < E_; ++e) attn[tid * E_ + e] = lg[e] * inv;
  }
  if (tid < C2_) {
    const float iv = gamma[tid] * rsqrtf(var[tid] + 1e-5f);
    bnscale[tid] = iv;
    bnbias[tid]  = beta[tid] - mean[tid] * iv;
  }
}

// --- Stage 3: combine experts with attention, pack bf16 A[b][c2][k] --------
__global__ __launch_bounds__(256) void pack_kernel(const float* __restrict__ ew,   // [E,C2,C1,3,3]
                                                   const float* __restrict__ attn, // [B,E]
                                                   unsigned short* __restrict__ Apack) {
  const int idx = blockIdx.x * 256 + threadIdx.x;   // [B][C2][KK_]
  const int b   = idx / (C2_ * KK_);
  const int rem = idx % (C2_ * KK_);
  const int c2  = rem / KK_;
  const int k   = rem % KK_;
  const int pos = k >> 6;                           // kh*3+kw
  const int c1  = k & 63;
  const size_t stride_e = (size_t)C2_ * C1_ * 9;
  const size_t w = ((size_t)c2 * C1_ + c1) * 9 + pos;
  float s = attn[b * E_ + 0] * ew[w]
          + attn[b * E_ + 1] * ew[w + stride_e]
          + attn[b * E_ + 2] * ew[w + 2 * stride_e]
          + attn[b * E_ + 3] * ew[w + 3 * stride_e];
  Apack[idx] = f2bf(s);
}

// ------------- Stage 4: implicit-GEMM conv via bf16 WMMA -------------------
__global__ __launch_bounds__(256) void dyconv_wmma_kernel(
    const float* __restrict__ x,
    const unsigned short* __restrict__ Apack,
    const float* __restrict__ bnscale,
    const float* __restrict__ bnbias,
    float* __restrict__ out) {
  // x halo tile, channels-last [r][c][ch], bf16: 6*34*64 = 13056 elems (26 KB)
  __shared__ __align__(16) unsigned short xs[HALO_H * HALO_W * C1_];

  const int b  = blockIdx.z;
  const int h0 = blockIdx.y * TILE_H;
  const int w0 = blockIdx.x * TILE_W;
  const int tid = threadIdx.x;

  // ---- load halo tile: coalesced global reads, convert f32 -> bf16 ----
  const int NTOT = HALO_H * HALO_W * C1_;
  for (int i = tid; i < NTOT; i += 256) {
    const int ch = i / (HALO_H * HALO_W);
    const int rr = i % (HALO_H * HALO_W);
    const int r  = rr / HALO_W;
    const int c  = rr % HALO_W;
    const int gh = h0 - 1 + r;
    const int gw = w0 - 1 + c;
    float v = 0.f;
    if ((unsigned)gh < (unsigned)H_ && (unsigned)gw < (unsigned)W_)
      v = x[(((size_t)b * C1_ + ch) * H_ + gh) * W_ + gw];
    xs[(r * HALO_W + c) * C1_ + ch] = f2bf(v);
  }
  __syncthreads();

  const int lane = tid & 31;
  const int wave = tid >> 5;
  const int mt   = wave & 3;        // c2 16-block
  const int ntg  = wave >> 2;       // n-tile group (4 tiles each)
  const int l16  = lane & 15;
  const bool hi  = lane >= 16;

  v8f acc[4] = {};

  int r_out[4], c_out[4];
  #pragma unroll
  for (int j = 0; j < 4; ++j) {
    const int p = (ntg * 4 + j) * 16 + l16;   // pixel 0..127 in tile
    r_out[j] = p >> 5;
    c_out[j] = p & 31;
  }

  // A fragment source: row-major [c2][k], 16B-aligned lane slices
  const unsigned short* Arow = Apack + ((size_t)b * C2_ + mt * 16 + l16) * KK_;
  const int aoff = hi ? 8  : 0;     // lanes 16-31 hold K {8..15, 24..31}
  const int boff = hi ? 8  : 0;     // lanes 16-31 hold K {8..15, 24..31}

  for (int kk = 0; kk < 18; ++kk) { // K = 576 = 18 * 32
    const int pos = kk >> 1;        // kh*3+kw
    const int kh  = pos / 3;
    const int kw  = pos - kh * 3;
    const int c1b = (kk & 1) * 32;

    FragU a;
    const unsigned short* ap = Arow + kk * 32 + aoff;
    a.u[0] = *reinterpret_cast<const v4u*>(ap);
    a.u[1] = *reinterpret_cast<const v4u*>(ap + 16);

    #pragma unroll
    for (int j = 0; j < 4; ++j) {
      FragU bb;
      const unsigned short* bp =
          &xs[((r_out[j] + kh) * HALO_W + (c_out[j] + kw)) * C1_ + c1b + boff];
      bb.u[0] = *reinterpret_cast<const v4u*>(bp);
      bb.u[1] = *reinterpret_cast<const v4u*>(bp + 16);
      acc[j] = __builtin_amdgcn_wmma_f32_16x16x32_bf16(
          false, a.v, false, bb.v, (short)0, acc[j], false, false);
    }
  }

  // ---- epilogue: BN + SiLU, per C/D VGPR layout ----
  #pragma unroll
  for (int j = 0; j < 4; ++j) {
    const int h = h0 + r_out[j];
    const int w = w0 + c_out[j];
    #pragma unroll
    for (int v = 0; v < 8; ++v) {
      const int c2 = mt * 16 + v + (hi ? 8 : 0);
      float o = acc[j][v] * bnscale[c2] + bnbias[c2];
      o = o * (1.0f / (1.0f + __expf(-o)));   // SiLU
      out[(((size_t)b * C2_ + c2) * H_ + h) * W_ + w] = o;
    }
  }
}

extern "C" void kernel_launch(void* const* d_in, const int* in_sizes, int n_in,
                              void* d_out, int out_size, void* d_ws, size_t ws_size,
                              hipStream_t stream) {
  (void)in_sizes; (void)n_in; (void)out_size; (void)ws_size;
  const float* x  = (const float*)d_in[0];
  const float* ew = (const float*)d_in[1];
  const float* w1 = (const float*)d_in[2];
  const float* w2 = (const float*)d_in[3];
  const float* g  = (const float*)d_in[4];
  const float* bt = (const float*)d_in[5];
  const float* mn = (const float*)d_in[6];
  const float* vr = (const float*)d_in[7];
  float* out = (float*)d_out;

  char* ws = (char*)d_ws;
  float* pooled  = (float*)(ws);                 // 1024 f32
  float* attn    = (float*)(ws + 4096);          // 64 f32
  float* bnscale = (float*)(ws + 4096 + 256);    // 64 f32
  float* bnbias  = (float*)(ws + 4096 + 512);    // 64 f32
  unsigned short* Apack = (unsigned short*)(ws + 8192);  // B*C2*KK bf16 = 1.13 MB

  pool_kernel<<<B_ * C1_, 256, 0, stream>>>(x, pooled);
  attn_bn_kernel<<<1, 64, 0, stream>>>(pooled, w1, w2, g, bt, mn, vr,
                                       attn, bnscale, bnbias);
  pack_kernel<<<(B_ * C2_ * KK_) / 256, 256, 0, stream>>>(ew, attn, Apack);

  dim3 grid(W_ / TILE_W, H_ / TILE_H, B_);       // 5 x 40 x 16
  dyconv_wmma_kernel<<<grid, 256, 0, stream>>>(x, Apack, bnscale, bnbias, out);
}
